// FourierFilter_32847909879913
// MI455X (gfx1250) — matrix operand order repaired and checked by reading, not compiled, not run on hardware
//
#include <hip/hip_runtime.h>
#include <hip/hip_bf16.h>

// Fourier high-pass filter via rank-1639 spectral projection, as two BF16 WMMA
// GEMMs with f32 accumulation:
//   coeff = X * [Cos;Sin]^T          (M=8192, N=1664, K=8192)
//   Y     = X - coeff_w * [Cos;Sin]  (M=8192, N=8192, K=1664)
// Block = 4 waves arranged 2x2 over a 128x128 tile. A/B panels (128x64) are
// staged in LDS with gfx1250 async global->LDS copies, double buffered on
// ASYNCcnt; one workgroup barrier per 64-wide K step (32 WMMAs/wave/step).

#define LN    8192          // sequence length (power of 2)
#define RROWS 8192          // B*C = 64*128
#define K0    820           // distinct |freq| bins zeroed (|f| < 10 Hz @ fs=100)
#define KP    832           // K0 padded to multiple of 16
#define NB    (2*KP)        // 1664 basis rows: [cos | sin]

typedef __bf16 bf16;
typedef __bf16 v16bf __attribute__((ext_vector_type(16)));
typedef __bf16 v8bf  __attribute__((ext_vector_type(8)));
typedef float  v8f   __attribute__((ext_vector_type(8)));

static __device__ __forceinline__ v16bf cat16(v8bf lo, v8bf hi) {
    return __builtin_shufflevector(lo, hi, 0,1,2,3,4,5,6,7,8,9,10,11,12,13,14,15);
}

// One async 16B copy per lane: global -> LDS (GV mode, tracked by ASYNCcnt).
static __device__ __forceinline__ void async_cp16(unsigned lds_off, const void* g) {
    asm volatile("global_load_async_to_lds_b128 %0, %1, off"
                 :: "v"(lds_off), "v"((unsigned long long)g) : "memory");
}
static __device__ __forceinline__ void wait_async0() {
    asm volatile("s_wait_asynccnt 0" ::: "memory");
}

// Stage one 128x64 A-panel and one 128x64 B-panel (row-major, 128B rows) into
// LDS: 2 x 16 KB = 2048 16B chunks, 16 async b128 ops per thread.
static __device__ __forceinline__ void stage_panels(
    const bf16* __restrict__ gA, size_t ldA,   // row r -> gA + r*ldA (kk folded in)
    const bf16* __restrict__ gB, size_t ldB,
    unsigned ldsA, unsigned ldsB, int t)
{
#pragma unroll
    for (int i = 0; i < 8; ++i) {
        int c = i * 128 + t;                    // chunk id 0..1023
        int row = c >> 3, seg = c & 7;          // 8 x 8-elem segs per 64-elem row
        async_cp16(ldsA + (unsigned)((row * 64 + seg * 8) * 2),
                   gA + (size_t)row * ldA + seg * 8);
    }
#pragma unroll
    for (int i = 0; i < 8; ++i) {
        int c = i * 128 + t;
        int row = c >> 3, seg = c & 7;
        async_cp16(ldsB + (unsigned)((row * 64 + seg * 8) * 2),
                   gB + (size_t)row * ldB + seg * 8);
    }
}

// ---------------------------------------------------------------------------
// Kernel 0: X (f32) -> Xb (bf16), streaming, 16 elements/thread.
// ---------------------------------------------------------------------------
__global__ void __launch_bounds__(256)
ff_cvt_x(const float* __restrict__ X, bf16* __restrict__ Xb)
{
    size_t i = ((size_t)blockIdx.x * blockDim.x + threadIdx.x) * 16;
    v8f lo = *(const v8f*)(X + i);
    v8f hi = *(const v8f*)(X + i + 8);
    v16bf o;
#pragma unroll
    for (int e = 0; e < 8; ++e) { o[e] = (bf16)lo[e]; o[8 + e] = (bf16)hi[e]; }
    *(v16bf*)(Xb + i) = o;
}

// ---------------------------------------------------------------------------
// Kernel 1: generate basis in bf16, both layouts.
//   Basis [NB][LN] : row j<KP -> cos(2*pi*j*n/L); row j>=KP -> sin(...)
//   BasisT[LN][NB] : transpose. Rows k in [K0,KP) are zero padding.
// ---------------------------------------------------------------------------
__global__ void __launch_bounds__(256)
ff_gen_basis(bf16* __restrict__ Basis, bf16* __restrict__ BasisT)
{
    int idx = blockIdx.x * blockDim.x + threadIdx.x;   // over KP*LN
    if (idx >= KP * LN) return;
    int k = idx >> 13;
    int n = idx & (LN - 1);
    float c = 0.f, s = 0.f;
    if (k < K0) {
        int t = (k * n) & (LN - 1);                    // exact index reduction
        float ang = (float)t * (6.28318530717958647692f / (float)LN);
        __sincosf(ang, &s, &c);
    }
    bf16 cb = (bf16)c, sb = (bf16)s;
    Basis [(size_t)k * LN + n]          = cb;
    Basis [(size_t)(KP + k) * LN + n]   = sb;
    BasisT[(size_t)n * NB + k]          = cb;
    BasisT[(size_t)n * NB + KP + k]     = sb;
}

// ---------------------------------------------------------------------------
// Kernel 2 (fast path): forward projection, LDS-staged.
// Cf[r][j] = w[j] * sum_n Xb[r][n]*Basis[j][n].  M=RROWS, N=NB, K=LN.
// Block tile 128x128 (2x2 waves x 64x64), K-steps of 64, double-buffered LDS.
// ---------------------------------------------------------------------------
__global__ void __launch_bounds__(128)
ff_fwd_bf(const bf16* __restrict__ Xb,     // [RROWS][LN]
          const bf16* __restrict__ Basis,  // [NB][LN]
          bf16*       __restrict__ Cf)     // [RROWS][NB]
{
    __shared__ bf16 As[2][128 * 64];
    __shared__ bf16 Bs[2][128 * 64];
    const int t    = threadIdx.x;
    const int lane = t & 31, wid = t >> 5;
    const int wr   = wid >> 1, wc = wid & 1;
    const int NBN  = NB / 128;                 // 13 block tiles along N
    const int bm   = blockIdx.x / NBN, bn = blockIdx.x % NBN;
    const int r0   = bm * 128, j0 = bn * 128;
    const int m    = lane & 15, h = lane >> 4;

    const unsigned ldsA0 = (unsigned)(unsigned long long)&As[0][0];
    const unsigned ldsA1 = (unsigned)(unsigned long long)&As[1][0];
    const unsigned ldsB0 = (unsigned)(unsigned long long)&Bs[0][0];
    const unsigned ldsB1 = (unsigned)(unsigned long long)&Bs[1][0];

    v8f acc[4][4];
#pragma unroll
    for (int i = 0; i < 4; ++i)
#pragma unroll
        for (int j = 0; j < 4; ++j) acc[i][j] = (v8f)(0.f);

    const bf16* gA = Xb    + (size_t)r0 * LN;
    const bf16* gB = Basis + (size_t)j0 * LN;

    stage_panels(gA, LN, gB, LN, ldsA0, ldsB0, t);       // prologue: step 0

    const int NS = LN / 64;                    // 128 steps
    for (int s = 0; s < NS; ++s) {
        wait_async0();          // my async copies for this step have landed
        __syncthreads();        // everyone's have
        const bf16* A  = As[s & 1];
        const bf16* Bp = Bs[s & 1];
        if (s + 1 < NS)
            stage_panels(gA + (size_t)(s + 1) * 64, LN,
                         gB + (size_t)(s + 1) * 64, LN,
                         (s & 1) ? ldsA0 : ldsA1,
                         (s & 1) ? ldsB0 : ldsB1, t);
#pragma unroll
        for (int kc = 0; kc < 64; kc += 32) {
            v16bf a[4], b[4];
#pragma unroll
            for (int mi = 0; mi < 4; ++mi) {
                const bf16* ar = A + (wr * 64 + mi * 16 + m) * 64 + kc;
                a[mi] = *(const v16bf*)(ar + h * 16);
            }
#pragma unroll
            for (int ni = 0; ni < 4; ++ni)
                b[ni] = *(const v16bf*)(Bp + (wc * 64 + ni * 16 + m) * 64 + kc + h * 16);
#pragma unroll
            for (int mi = 0; mi < 4; ++mi)
#pragma unroll
                for (int ni = 0; ni < 4; ++ni)
                    acc[mi][ni] = __builtin_amdgcn_wmma_f32_16x16x32_bf16(
                        false, a[mi], false, b[ni], (short)0, acc[mi][ni], false, false);
        }
    }

    const float invL = 1.0f / (float)LN;
#pragma unroll
    for (int ni = 0; ni < 4; ++ni) {
        int j    = j0 + wc * 64 + ni * 16 + m;
        int kbin = j % KP;
        float wgt = (kbin == 0 ? 1.0f : 2.0f) * invL;    // 1/L for DC, 2/L otherwise
#pragma unroll
        for (int mi = 0; mi < 4; ++mi)
#pragma unroll
            for (int v = 0; v < 8; ++v) {
                size_t row = (size_t)(r0 + wr * 64 + mi * 16 + v + 8 * h);
                Cf[row * NB + j] = (bf16)(acc[mi][ni][v] * wgt);
            }
    }
}

// ---------------------------------------------------------------------------
// Kernel 2 (fallback): forward projection with inline f32->bf16 conversion,
// direct global fragment loads; used only if ws can't hold bf16 X.
// ---------------------------------------------------------------------------
__global__ void __launch_bounds__(128)
ff_fwd_cvt(const float* __restrict__ X,
           const bf16*  __restrict__ Basis,
           bf16*        __restrict__ Cf)
{
    const int lane = threadIdx.x & 31;
    const int wid  = threadIdx.x >> 5;
    const int w    = blockIdx.x * 4 + wid;
    const int NWN  = NB / 64;
    const int wm   = w / NWN, wn = w % NWN;
    const int r0   = wm * 64, j0 = wn * 64;
    const int m    = lane & 15;
    const int h    = lane >> 4;

    v8f acc[4][4];
#pragma unroll
    for (int i = 0; i < 4; ++i)
#pragma unroll
        for (int j = 0; j < 4; ++j) acc[i][j] = (v8f)(0.f);

    for (int kk = 0; kk < LN; kk += 32) {
        v16bf a[4], b[4];
#pragma unroll
        for (int mi = 0; mi < 4; ++mi) {
            const float* xr = X + (size_t)(r0 + mi * 16 + m) * LN + kk;
            v8f lo = *(const v8f*)(xr + h * 16);
            v8f hi = *(const v8f*)(xr + h * 16 + 8);
#pragma unroll
            for (int e = 0; e < 8; ++e) { a[mi][e] = (bf16)lo[e]; a[mi][8 + e] = (bf16)hi[e]; }
        }
#pragma unroll
        for (int ni = 0; ni < 4; ++ni)
            b[ni] = *(const v16bf*)(Basis + (size_t)(j0 + ni * 16 + m) * LN + kk + h * 16);
#pragma unroll
        for (int mi = 0; mi < 4; ++mi)
#pragma unroll
            for (int ni = 0; ni < 4; ++ni)
                acc[mi][ni] = __builtin_amdgcn_wmma_f32_16x16x32_bf16(
                    false, a[mi], false, b[ni], (short)0, acc[mi][ni], false, false);
    }

    const float invL = 1.0f / (float)LN;
#pragma unroll
    for (int ni = 0; ni < 4; ++ni) {
        int j    = j0 + ni * 16 + m;
        int kbin = j % KP;
        float wgt = (kbin == 0 ? 1.0f : 2.0f) * invL;
#pragma unroll
        for (int mi = 0; mi < 4; ++mi)
#pragma unroll
            for (int v = 0; v < 8; ++v) {
                size_t row = (size_t)(r0 + mi * 16 + v + 8 * h);
                Cf[row * NB + j] = (bf16)(acc[mi][ni][v] * wgt);
            }
    }
}

// ---------------------------------------------------------------------------
// Kernel 3: inverse projection + subtract, LDS-staged.
//   Y[r][n] = X[r][n] - sum_j Cf[r][j]*Basis[j][n]
// M=RROWS, N=LN, K=NB (26 steps of 64). Block tile 128x128 (2x2 waves).
// ---------------------------------------------------------------------------
__global__ void __launch_bounds__(128)
ff_inv(const float* __restrict__ X,       // [RROWS][LN] (f32, residual)
       const bf16*  __restrict__ Cf,      // [RROWS][NB]
       const bf16*  __restrict__ BasisT,  // [LN][NB]
       float*       __restrict__ Y)       // [RROWS][LN]
{
    __shared__ bf16 As[2][128 * 64];
    __shared__ bf16 Bs[2][128 * 64];
    const int t    = threadIdx.x;
    const int lane = t & 31, wid = t >> 5;
    const int wr   = wid >> 1, wc = wid & 1;
    const int NBN  = LN / 128;                 // 64 block tiles along N
    const int bm   = blockIdx.x / NBN, bn = blockIdx.x % NBN;
    const int r0   = bm * 128, n0 = bn * 128;
    const int m    = lane & 15, h = lane >> 4;

    const unsigned ldsA0 = (unsigned)(unsigned long long)&As[0][0];
    const unsigned ldsA1 = (unsigned)(unsigned long long)&As[1][0];
    const unsigned ldsB0 = (unsigned)(unsigned long long)&Bs[0][0];
    const unsigned ldsB1 = (unsigned)(unsigned long long)&Bs[1][0];

    v8f acc[4][4];
#pragma unroll
    for (int i = 0; i < 4; ++i)
#pragma unroll
        for (int j = 0; j < 4; ++j) acc[i][j] = (v8f)(0.f);

    const bf16* gA = Cf     + (size_t)r0 * NB;
    const bf16* gB = BasisT + (size_t)n0 * NB;

    stage_panels(gA, NB, gB, NB, ldsA0, ldsB0, t);

    const int NS = NB / 64;                    // 26 steps
    for (int s = 0; s < NS; ++s) {
        wait_async0();
        __syncthreads();
        const bf16* A  = As[s & 1];
        const bf16* Bp = Bs[s & 1];
        if (s + 1 < NS)
            stage_panels(gA + (size_t)(s + 1) * 64, NB,
                         gB + (size_t)(s + 1) * 64, NB,
                         (s & 1) ? ldsA0 : ldsA1,
                         (s & 1) ? ldsB0 : ldsB1, t);
#pragma unroll
        for (int kc = 0; kc < 64; kc += 32) {
            v16bf a[4], b[4];
#pragma unroll
            for (int mi = 0; mi < 4; ++mi) {
                const bf16* ar = A + (wr * 64 + mi * 16 + m) * 64 + kc;
                a[mi] = *(const v16bf*)(ar + h * 16);
            }
#pragma unroll
            for (int ni = 0; ni < 4; ++ni)
                b[ni] = *(const v16bf*)(Bp + (wc * 64 + ni * 16 + m) * 64 + kc + h * 16);
#pragma unroll
            for (int mi = 0; mi < 4; ++mi)
#pragma unroll
                for (int ni = 0; ni < 4; ++ni)
                    acc[mi][ni] = __builtin_amdgcn_wmma_f32_16x16x32_bf16(
                        false, a[mi], false, b[ni], (short)0, acc[mi][ni], false, false);
        }
    }

    // Epilogue: y = x - projection
#pragma unroll
    for (int mi = 0; mi < 4; ++mi)
#pragma unroll
        for (int v = 0; v < 8; ++v) {
            size_t row = (size_t)(r0 + wr * 64 + mi * 16 + v + 8 * h);
#pragma unroll
            for (int ni = 0; ni < 4; ++ni) {
                size_t col = (size_t)(n0 + wc * 64 + ni * 16 + m);
                Y[row * LN + col] = X[row * LN + col] - acc[mi][ni][v];
            }
        }
}

// ---------------------------------------------------------------------------
extern "C" void kernel_launch(void* const* d_in, const int* in_sizes, int n_in,
                              void* d_out, int out_size, void* d_ws, size_t ws_size,
                              hipStream_t stream) {
    (void)in_sizes; (void)n_in; (void)out_size;
    const float* X = (const float*)d_in[0];
    float*       Y = (float*)d_out;

    const size_t basisBytes = (size_t)NB * LN * sizeof(bf16);      // ~26 MB
    const size_t cfBytes    = (size_t)RROWS * NB * sizeof(bf16);   // ~26 MB
    const size_t xbBytes    = (size_t)RROWS * LN * sizeof(bf16);   // 128 MB
    const size_t needSmall  = 2 * basisBytes + cfBytes;            // ~78 MB
    const size_t needFull   = needSmall + xbBytes;                 // ~206 MB
    if (ws_size < needSmall) return;

    bf16* Basis  = (bf16*)d_ws;
    bf16* BasisT = (bf16*)((char*)d_ws + basisBytes);
    bf16* Cf     = (bf16*)((char*)d_ws + 2 * basisBytes);
    bf16* Xb     = (bf16*)((char*)d_ws + needSmall);

    // 1) basis generation
    ff_gen_basis<<<(KP * LN + 255) / 256, 256, 0, stream>>>(Basis, BasisT);

    // 2) forward GEMM
    if (ws_size >= needFull) {
        ff_cvt_x<<<(int)(((size_t)RROWS * LN / 16) / 256), 256, 0, stream>>>(X, Xb);
        ff_fwd_bf<<<(RROWS / 128) * (NB / 128), 128, 0, stream>>>(Xb, Basis, Cf);
    } else {
        ff_fwd_cvt<<<(128 * (NB / 64)) / 4, 128, 0, stream>>>(X, Basis, Cf);
    }

    // 3) inverse GEMM + subtract
    ff_inv<<<(RROWS / 128) * (LN / 128), 128, 0, stream>>>(X, Cf, BasisT, Y);
}
